// YuanMoeLayer_3332894622533
// MI455X (gfx1250) — matrix operand order repaired and checked by reading, not compiled, not run on hardware
//
#include <hip/hip_runtime.h>
#include <math.h>

typedef __bf16 bf16;
typedef __attribute__((ext_vector_type(16))) __bf16 v16bf;
typedef __attribute__((ext_vector_type(8)))  __bf16 v8bf;
typedef __attribute__((ext_vector_type(8)))  float  v8f;
typedef __attribute__((ext_vector_type(4)))  int    v4i;

constexpr int T_T  = 2048;   // tokens
constexpr int H_H  = 2048;   // hidden
constexpr int E_E  = 8;      // experts
constexpr int I_I  = 4096;   // ffn inner
constexpr int LDA  = 40;     // padded LDS row stride in bf16 elems (80B, conflict-free)

#define AS1 __attribute__((address_space(1)))
#define AS3 __attribute__((address_space(3)))

// ---- CDNA5 async Global->LDS path (guarded; falls back to reg prefetch) ----
#if defined(__gfx1250__) && __has_builtin(__builtin_amdgcn_global_load_async_to_lds_b128)
#define HAVE_ASYNC_LDS 1
#else
#define HAVE_ASYNC_LDS 0
#endif

#if defined(__gfx1250__) && __has_builtin(__builtin_amdgcn_s_wait_asynccnt)
#define WAIT_ASYNC() __builtin_amdgcn_s_wait_asynccnt(0)
#elif defined(__gfx1250__)
#define WAIT_ASYNC() asm volatile("s_wait_asynccnt 0x0" ::: "memory")
#else
#define WAIT_ASYNC()
#endif

#if HAVE_ASYNC_LDS
__device__ __forceinline__ void async_cp16(__bf16* dst_lds, const bf16* src_global) {
  AS1 v4i* s = (AS1 v4i*)(unsigned long long)dst_lds == 0
                   ? (AS1 v4i*)0
                   : (AS1 v4i*)(unsigned long long)src_global;  // keep simple cast below
  (void)s;
  AS1 v4i* sp = (AS1 v4i*)(unsigned long long)src_global;
  AS3 v4i* dp = (AS3 v4i*)(unsigned)(unsigned long long)dst_lds;
  __builtin_amdgcn_global_load_async_to_lds_b128(sp, dp, 0, 0);
}
#endif

// ---------------- workspace layout ----------------
// [0,32)                        counts[8]
// [64,100)                      offs[9]
// [128, 128+E*T*4)              toklist (packed t*2+k)
// [+E*T*4, +E*T*8)              gates (float per (e,pos))
// [128+E*T*8, ...)              act bf16 [T*2][I]   (~33.5 MB)

__device__ __forceinline__ v8bf cvt_v8(const float4 a, const float4 b) {
  v8bf r;
  r[0] = (__bf16)a.x; r[1] = (__bf16)a.y; r[2] = (__bf16)a.z; r[3] = (__bf16)a.w;
  r[4] = (__bf16)b.x; r[5] = (__bf16)b.y; r[6] = (__bf16)b.z; r[7] = (__bf16)b.w;
  return r;
}

__device__ __forceinline__ v16bf frag16(const __bf16* p0, const __bf16* p1) {
  v8bf a = *(const v8bf*)p0;
  v8bf b = *(const v8bf*)p1;
  return __builtin_shufflevector(a, b, 0,1,2,3,4,5,6,7,8,9,10,11,12,13,14,15);
}

// ---------------- init: zero output + per-expert counts ----------------
__global__ void k_init(float* __restrict__ out, int* __restrict__ counts) {
  size_t i = (size_t)blockIdx.x * blockDim.x + threadIdx.x;
  size_t stride = (size_t)gridDim.x * blockDim.x;
  for (size_t j = i; j < (size_t)T_T * H_H; j += stride) out[j] = 0.f;
  if (i < E_E) counts[i] = 0;
}

// ---------------- router: mix -> attn softmax over experts -> top2 -> scatter ----
__global__ __launch_bounds__(256)
void k_router(const float* __restrict__ x, const float* __restrict__ wr,
              int* __restrict__ counts, int* __restrict__ toklist,
              float* __restrict__ gates) {
  const int lane = threadIdx.x & 31;
  const int wid  = threadIdx.x >> 5;
  const int t = blockIdx.x * 8 + wid;     // one wave per token

  float acc[24];
  #pragma unroll
  for (int n = 0; n < 24; ++n) acc[n] = 0.f;

  const float* xr = x + (size_t)t * H_H;
  for (int h = lane * 4; h < H_H; h += 128) {
    float4 xv = *(const float4*)(xr + h);
    #pragma unroll
    for (int n = 0; n < 24; ++n) {
      float4 wv = *(const float4*)(wr + (size_t)n * H_H + h);
      acc[n] += xv.x * wv.x + xv.y * wv.y + xv.z * wv.z + xv.w * wv.w;
    }
  }
  #pragma unroll
  for (int n = 0; n < 24; ++n) {
    #pragma unroll
    for (int off = 16; off > 0; off >>= 1)
      acc[n] += __shfl_xor(acc[n], off, 32);
  }

  if (lane == 0) {
    // q = acc[0..7], k = acc[8..15], v = acc[16..23]
    float lg[E_E];
    #pragma unroll
    for (int i = 0; i < E_E; ++i) {
      float q = acc[i];
      float m = -INFINITY;
      #pragma unroll
      for (int j = 0; j < E_E; ++j) m = fmaxf(m, q * acc[8 + j]);
      float se = 0.f, sv = 0.f;
      #pragma unroll
      for (int j = 0; j < E_E; ++j) {
        float eij = __expf(q * acc[8 + j] - m);
        se += eij; sv += eij * acc[16 + j];
      }
      lg[i] = sv * __builtin_amdgcn_rcpf(se);
    }
    int i0 = 0; float l0 = lg[0];
    #pragma unroll
    for (int i = 1; i < E_E; ++i) if (lg[i] > l0) { l0 = lg[i]; i0 = i; }
    int i1 = -1; float l1 = -INFINITY;
    #pragma unroll
    for (int i = 0; i < E_E; ++i) if (i != i0 && lg[i] > l1) { l1 = lg[i]; i1 = i; }
    float ex = __expf(l1 - l0);
    float p0 = __builtin_amdgcn_rcpf(1.f + ex);
    float p1 = ex * p0;
    int pos0 = atomicAdd(&counts[i0], 1);
    toklist[i0 * T_T + pos0] = t * 2 + 0;
    gates[i0 * T_T + pos0]   = p0;
    int pos1 = atomicAdd(&counts[i1], 1);
    toklist[i1 * T_T + pos1] = t * 2 + 1;
    gates[i1 * T_T + pos1]   = p1;
  }
}

// ---------------- tiny exclusive scan over 8 experts ----------------
__global__ void k_offsets(const int* __restrict__ counts, int* __restrict__ offs) {
  if (blockIdx.x == 0 && threadIdx.x == 0) {
    int s = 0;
    for (int e = 0; e < E_E; ++e) { offs[e] = s; s += counts[e]; }
    offs[E_E] = s;
  }
}

// ---------------- GEMM1: act = silu(x@w1_g^T) * (x@w1_u^T) ----------------
// gathered rows, fp32->bf16 on the fly, register double-buffer + LDS ping-pong
__global__ __launch_bounds__(256)
void k_gemm1(const float* __restrict__ x, const float* __restrict__ w1,
             const int* __restrict__ counts, const int* __restrict__ offs,
             const int* __restrict__ toklist, bf16* __restrict__ act) {
  const int e = blockIdx.z;
  const int cnt = counts[e];
  const int mBase = blockIdx.y * 128;
  if (mBase >= cnt) return;
  const int iBase = blockIdx.x * 64;

  __shared__ __bf16 As[2][128 * LDA];
  __shared__ __bf16 Bg[2][64 * LDA];
  __shared__ __bf16 Bu[2][64 * LDA];

  const int tid  = threadIdx.x;
  const int wid  = tid >> 5;
  const int lane = tid & 31;
  const int wm = wid >> 1;      // 0..3  (M quadrant of 32)
  const int wn = wid & 1;       // 0..1  (N half of 32)
  const int lm = lane & 15;
  const int lg = lane >> 4;

  // A staging: 2 threads/row, 16 floats each
  const int arow = tid >> 1;
  const int akp  = (tid & 1) << 4;
  int gposA = mBase + arow;
  int tok = 0;
  if (gposA < cnt) tok = toklist[e * T_T + gposA] >> 1;
  const float* aSrc = x + (size_t)tok * H_H + akp;

  // B staging: 128 threads gate, 128 threads up; 2 threads/row
  const int bhalf = tid >> 7;
  const int bt    = tid & 127;
  const int brow  = bt >> 1;
  const int bkp   = (bt & 1) << 4;
  const size_t ncol = (size_t)(bhalf ? (I_I + iBase + brow) : (iBase + brow));
  const float* bSrc = w1 + ((size_t)e * 2 * I_I + ncol) * H_H + bkp;

  float4 pa[4], pb[4];
  auto loadG = [&](int k0) {
    const float4* a4 = (const float4*)(aSrc + k0);
    pa[0] = a4[0]; pa[1] = a4[1]; pa[2] = a4[2]; pa[3] = a4[3];
    const float4* b4 = (const float4*)(bSrc + k0);
    pb[0] = b4[0]; pb[1] = b4[1]; pb[2] = b4[2]; pb[3] = b4[3];
    __builtin_prefetch(aSrc + k0 + 64, 0, 3);   // L2/WGP$ warmup 2 steps ahead
    __builtin_prefetch(bSrc + k0 + 64, 0, 3);
  };

  v8f accg[2][2] = {};
  v8f accu[2][2] = {};

  constexpr int NIT = H_H / 32;
  loadG(0);
  for (int it = 0; it < NIT; ++it) {
    const int sel = it & 1;
    // commit prefetched regs to LDS[sel]
    *(v8bf*)(As[sel] + arow * LDA + akp)     = cvt_v8(pa[0], pa[1]);
    *(v8bf*)(As[sel] + arow * LDA + akp + 8) = cvt_v8(pa[2], pa[3]);
    __bf16* bDst = (bhalf ? Bu[sel] : Bg[sel]) + brow * LDA + bkp;
    *(v8bf*)(bDst)     = cvt_v8(pb[0], pb[1]);
    *(v8bf*)(bDst + 8) = cvt_v8(pb[2], pb[3]);
    __syncthreads();

    if (it + 1 < NIT) loadG((it + 1) * 32);   // overlap next global loads w/ WMMA

    v16bf af[2], bgf[2], buf_[2];
    #pragma unroll
    for (int s = 0; s < 2; ++s) {
      const __bf16* p = As[sel] + (wm * 32 + s * 16 + lm) * LDA;
      af[s] = frag16(p + lg * 16, p + lg * 16 + 8);
    }
    #pragma unroll
    for (int u = 0; u < 2; ++u) {
      const __bf16* pg = Bg[sel] + (wn * 32 + u * 16 + lm) * LDA + lg * 16;
      bgf[u] = frag16(pg, pg + 8);
      const __bf16* pu = Bu[sel] + (wn * 32 + u * 16 + lm) * LDA + lg * 16;
      buf_[u] = frag16(pu, pu + 8);
    }
    #pragma unroll
    for (int s = 0; s < 2; ++s) {
      #pragma unroll
      for (int u = 0; u < 2; ++u) {
        accg[s][u] = __builtin_amdgcn_wmma_f32_16x16x32_bf16(
            false, af[s], false, bgf[u], (short)0, accg[s][u], false, false);
        accu[s][u] = __builtin_amdgcn_wmma_f32_16x16x32_bf16(
            false, af[s], false, buf_[u], (short)0, accu[s][u], false, false);
      }
    }
  }

  const int rowGlobalBase = offs[e] + mBase;
  #pragma unroll
  for (int s = 0; s < 2; ++s) {
    #pragma unroll
    for (int u = 0; u < 2; ++u) {
      const int colLocal = wn * 32 + u * 16 + lm;
      #pragma unroll
      for (int r = 0; r < 8; ++r) {
        int rowLocal = wm * 32 + s * 16 + lg * 8 + r;
        if (mBase + rowLocal < cnt) {
          float g  = accg[s][u][r];
          float up = accu[s][u][r];
          float sig = __builtin_amdgcn_rcpf(1.f + __expf(-g));  // v_rcp_f32
          float a = g * sig * up;                                // silu(g) * up
          act[(size_t)(rowGlobalBase + rowLocal) * I_I + iBase + colLocal] = (__bf16)a;
        }
      }
    }
  }
}

// ---------------- GEMM2: out[t] += gate * (act @ w2^T), deterministic 2-way atomics ----
// A-operand is bf16: staged with async Global->LDS copies (ASYNCcnt) when available.
__global__ __launch_bounds__(256)
void k_gemm2(const bf16* __restrict__ act, const float* __restrict__ w2,
             const int* __restrict__ counts, const int* __restrict__ offs,
             const int* __restrict__ toklist, const float* __restrict__ gates,
             float* __restrict__ out) {
  const int e = blockIdx.z;
  const int cnt = counts[e];
  const int mBase = blockIdx.y * 128;
  if (mBase >= cnt) return;
  const int hBase = blockIdx.x * 64;

  __shared__ __bf16 As[2][128 * LDA];
  __shared__ __bf16 Bs[2][64 * LDA];

  const int tid  = threadIdx.x;
  const int wid  = tid >> 5;
  const int lane = tid & 31;
  const int wm = wid >> 1;
  const int wn = wid & 1;
  const int lm = lane & 15;
  const int lg = lane >> 4;

  // A staging: already bf16, 2 threads/row, 16 halves each
  const int arow = tid >> 1;
  const int akp  = (tid & 1) << 4;
  int gposA = mBase + arow;
  if (gposA >= cnt) gposA = cnt - 1;          // clamp (rows discarded in epilogue)
  const bf16* aSrc = act + (size_t)(offs[e] + gposA) * I_I + akp;

  // B staging: 4 threads/row, 8 floats each
  const int brow = tid >> 2;
  const int bkp  = (tid & 3) << 3;
  const float* bSrc = w2 + ((size_t)e * H_H + hBase + brow) * I_I + bkp;

  float4 pB[2];
  auto loadB = [&](int k0) {
    const float4* b4 = (const float4*)(bSrc + k0);
    pB[0] = b4[0]; pB[1] = b4[1];
    __builtin_prefetch(bSrc + k0 + 64, 0, 3);
  };

#if HAVE_ASYNC_LDS
  auto issueA = [&](int sel, int k0) {
    async_cp16(As[sel] + arow * LDA + akp,     aSrc + k0);
    async_cp16(As[sel] + arow * LDA + akp + 8, aSrc + k0 + 8);
  };
#else
  v8bf pA[2];
  auto loadA = [&](int k0) {
    pA[0] = *(const v8bf*)(aSrc + k0);
    pA[1] = *(const v8bf*)(aSrc + k0 + 8);
  };
#endif

  v8f acc[2][2] = {};

  constexpr int NIT = I_I / 32;
#if HAVE_ASYNC_LDS
  issueA(0, 0);
#else
  loadA(0);
#endif
  loadB(0);
  for (int it = 0; it < NIT; ++it) {
    const int sel = it & 1;
#if !HAVE_ASYNC_LDS
    *(v8bf*)(As[sel] + arow * LDA + akp)     = pA[0];
    *(v8bf*)(As[sel] + arow * LDA + akp + 8) = pA[1];
#endif
    *(v8bf*)(Bs[sel] + brow * LDA + bkp)     = cvt_v8(pB[0], pB[1]);
#if HAVE_ASYNC_LDS
    WAIT_ASYNC();                  // A[sel] async copies complete (per-wave ASYNCcnt)
#endif
    __syncthreads();               // publish A[sel] + B[sel] to all waves

    if (it + 1 < NIT) {
#if HAVE_ASYNC_LDS
      issueA(sel ^ 1, (it + 1) * 32);   // buffer sel^1 free: last read in previous iter, before this barrier
#else
      loadA((it + 1) * 32);
#endif
      loadB((it + 1) * 32);
    }

    v16bf af[2], bf_[2];
    #pragma unroll
    for (int s = 0; s < 2; ++s) {
      const __bf16* p = As[sel] + (wm * 32 + s * 16 + lm) * LDA;
      af[s] = frag16(p + lg * 16, p + lg * 16 + 8);
    }
    #pragma unroll
    for (int u = 0; u < 2; ++u) {
      const __bf16* pb = Bs[sel] + (wn * 32 + u * 16 + lm) * LDA + lg * 16;
      bf_[u] = frag16(pb, pb + 8);
    }
    #pragma unroll
    for (int s = 0; s < 2; ++s)
      #pragma unroll
      for (int u = 0; u < 2; ++u)
        acc[s][u] = __builtin_amdgcn_wmma_f32_16x16x32_bf16(
            false, af[s], false, bf_[u], (short)0, acc[s][u], false, false);
  }

  #pragma unroll
  for (int s = 0; s < 2; ++s) {
    #pragma unroll
    for (int u = 0; u < 2; ++u) {
      const int colLocal = wn * 32 + u * 16 + lm;
      #pragma unroll
      for (int r = 0; r < 8; ++r) {
        int rowLocal = wm * 32 + s * 16 + lg * 8 + r;
        int pos = mBase + rowLocal;
        if (pos < cnt) {
          int   t  = toklist[e * T_T + pos] >> 1;
          float gv = gates[e * T_T + pos];
          atomicAdd(out + (size_t)t * H_H + hBase + colLocal, gv * acc[s][u][r]);
        }
      }
    }
  }
}

extern "C" void kernel_launch(void* const* d_in, const int* in_sizes, int n_in,
                              void* d_out, int out_size, void* d_ws, size_t ws_size,
                              hipStream_t stream) {
  (void)in_sizes; (void)n_in; (void)out_size; (void)ws_size;
  const float* x  = (const float*)d_in[0];   // [T,H]
  const float* wr = (const float*)d_in[1];   // [3E,H]
  const float* w1 = (const float*)d_in[2];   // [E,2I,H]
  const float* w2 = (const float*)d_in[3];   // [E,H,I]
  float* out = (float*)d_out;                // [T,H]

  char* ws = (char*)d_ws;
  int*   counts  = (int*)ws;                           // 8 ints
  int*   offs    = (int*)(ws + 64);                    // 9 ints
  int*   toklist = (int*)(ws + 128);                   // E*T ints
  float* gates   = (float*)(ws + 128 + (size_t)E_E * T_T * 4);
  bf16*  act     = (bf16*)(ws + 128 + (size_t)E_E * T_T * 8);   // [T*2][I] bf16

  k_init<<<512, 256, 0, stream>>>(out, counts);
  k_router<<<T_T / 8, 256, 0, stream>>>(x, wr, counts, toklist, gates);
  k_offsets<<<1, 32, 0, stream>>>(counts, offs);
  k_gemm1<<<dim3(I_I / 64, T_T / 128, E_E), 256, 0, stream>>>(
      x, w1, counts, offs, toklist, act);
  k_gemm2<<<dim3(H_H / 64, T_T / 128, E_E), 256, 0, stream>>>(
      act, w2, counts, offs, toklist, gates, out);
}
